// CorrectedGNNVectorField_12214886990296
// MI455X (gfx1250) — matrix operand order repaired and checked by reading, not compiled, not run on hardware
//
#include <hip/hip_runtime.h>
#include <hip/hip_bf16.h>
#include <math.h>

typedef __attribute__((ext_vector_type(16))) _Float16 v16h;
typedef __attribute__((ext_vector_type(8)))  float    v8f;
typedef __attribute__((ext_vector_type(4)))  unsigned int u32x4_t;
typedef __attribute__((ext_vector_type(8)))  int      i32x8_t;
typedef __attribute__((ext_vector_type(4)))  int      i32x4_t;

#define N_LATENT 50
#define N_GENES  20000
#define N_COND   10
#define N_HID    64
#define HEADS    2
#define BATCH    4096
#define N_EDGES  640000
#define D_OUT    128
#define E_ALL    (N_EDGES + N_GENES)   // edges + self loops

// ---------------------------------------------------------------- utilities
__device__ __forceinline__ float siluf(float x) { return x / (1.f + __expf(-x)); }

// float atomic max via signed/unsigned int trick (lowers to global_atomic_max/min)
__device__ __forceinline__ void atomicMaxFloat(float* addr, float val) {
  if (val >= 0.f) atomicMax((int*)addr, __float_as_int(val));
  else            atomicMin((unsigned int*)addr, __float_as_uint(val));
}

// ---- Tensor Data Mover: DMA one 32-gene x 128-ch f16 tile (8 KB) into LDS.
// D# per CDNA5 ISA ch.8: group0 = {count, lds_addr, global_addr[56:0], type=2},
// group1 = dims/strides in data_size units (data_size=3 -> 8-byte units, row=32u).
__device__ __forceinline__ void tdm_load_tile(unsigned lds_addr, const _Float16* gsrc) {
  unsigned long long ga = (unsigned long long)(uintptr_t)gsrc;
  u32x4_t d0;
  d0.x = 1u;                                                 // count=1 (valid), user mode
  d0.y = lds_addr;                                           // bits 63:32 lds_addr
  d0.z = (unsigned)(ga & 0xffffffffull);                     // global_addr[31:0]
  d0.w = (unsigned)((ga >> 32) & 0x1ffffffull) | (2u << 30); // [56:32] + type=2 ("image")
  i32x8_t d1;
  d1[0] = 0x30000;                 // wg_mask=0 | data_size=3 (8B units)
  d1[1] = 32 << 16;                // tensor_dim0 = 32 units (256 B row), bits 79:48
  d1[2] = (int)(20000u << 16);     // tensor_dim1 = 20000 rows, bits 111:80
  d1[3] = 32 << 16;                // tile_dim0 = 32 units, bits 127:112
  d1[4] = 32;                      // tile_dim1 = 32 rows; tile_dim2 = 0
  d1[5] = 32;                      // tensor_dim0_stride = 32 units (48-bit field)
  d1[6] = 0;
  d1[7] = 0;                       // tensor_dim1_stride = 0 (2D)
  i32x4_t z4 = {0, 0, 0, 0};
#if __clang_major__ >= 23
  i32x8_t z8 = {0, 0, 0, 0, 0, 0, 0, 0};
  __builtin_amdgcn_tensor_load_to_lds(d0, d1, z4, z4, z8, 0);
#else
  __builtin_amdgcn_tensor_load_to_lds(d0, d1, z4, z4, 0);
#endif
}

// ---------------------------------------------------------------- GATv2 pieces
__global__ __launch_bounds__(256)
void k_node_transform(const float* __restrict__ ge,
                      const float* __restrict__ Wl, const float* __restrict__ bl,
                      const float* __restrict__ Wr, const float* __restrict__ br,
                      float* __restrict__ xl, float* __restrict__ xr) {
  int i = blockIdx.x * blockDim.x + threadIdx.x;
  if (i >= N_GENES * D_OUT) return;
  int n = i / D_OUT, d = i % D_OUT;
  const float* g = ge + (size_t)n * N_HID;
  float al = bl[d], ar = br[d];
  for (int h = 0; h < N_HID; ++h) {
    float gv = g[h];
    al += gv * Wl[h * D_OUT + d];
    ar += gv * Wr[h * D_OUT + d];
  }
  xl[i] = al; xr[i] = ar;
}

__global__ __launch_bounds__(256)
void k_init(float* __restrict__ accum, float* __restrict__ emax, float* __restrict__ den) {
  int i = blockIdx.x * blockDim.x + threadIdx.x;
  if (i < N_GENES * D_OUT) accum[i] = 0.f;
  if (i < N_GENES * HEADS) { ((int*)emax)[i] = 0xFF800000; den[i] = 0.f; }  // -inf
}

__global__ __launch_bounds__(256)
void k_edge_score(const int* __restrict__ ei,
                  const float* __restrict__ xl, const float* __restrict__ xr,
                  const float* __restrict__ att,
                  float* __restrict__ escore, float* __restrict__ emax) {
  int tid = blockIdx.x * blockDim.x + threadIdx.x;
  if (tid >= E_ALL * HEADS) return;
  int e = tid >> 1, h = tid & 1;
  int src, dst;
  if (e < N_EDGES) { src = ei[e]; dst = ei[N_EDGES + e]; }
  else             { src = dst = e - N_EDGES; }            // self loop
  const float* pl = xl + (size_t)src * D_OUT + h * N_HID;
  const float* pr = xr + (size_t)dst * D_OUT + h * N_HID;
  const float* pa = att + h * N_HID;
  float s = 0.f;
  for (int c = 0; c < N_HID; ++c) {
    float m = pl[c] + pr[c];
    float lr = m > 0.f ? m : 0.2f * m;
    s += lr * pa[c];
  }
  escore[e * 2 + h] = s;
  atomicMaxFloat(&emax[dst * 2 + h], s);
}

__global__ __launch_bounds__(256)
void k_edge_accum(const int* __restrict__ ei,
                  const float* __restrict__ xl,
                  const float* __restrict__ escore, const float* __restrict__ emax,
                  float* __restrict__ den, float* __restrict__ accum) {
  int tid = blockIdx.x * blockDim.x + threadIdx.x;
  if (tid >= E_ALL * HEADS) return;
  int e = tid >> 1, h = tid & 1;
  int src, dst;
  if (e < N_EDGES) { src = ei[e]; dst = ei[N_EDGES + e]; }
  else             { src = dst = e - N_EDGES; }
  float ex = __expf(escore[e * 2 + h] - emax[dst * 2 + h]);
  atomicAdd(&den[dst * 2 + h], ex);
  const float* pl = xl + (size_t)src * D_OUT + h * N_HID;
  float* pa = accum + (size_t)dst * D_OUT + h * N_HID;
  for (int c = 0; c < N_HID; ++c) atomicAdd(&pa[c], ex * pl[c]);
}

__global__ __launch_bounds__(256)
void k_finalize(const float* __restrict__ accum, const float* __restrict__ den,
                const float* __restrict__ gbias, _Float16* __restrict__ gfh) {
  int i = blockIdx.x * blockDim.x + threadIdx.x;
  if (i >= N_GENES * D_OUT) return;
  int n = i / D_OUT, d = i % D_OUT;
  gfh[i] = (_Float16)(accum[i] / den[n * 2 + (d / N_HID)] + gbias[d]);
}

// ---------------------------------------------------------------- query + small embeds
__global__ __launch_bounds__(256)
void k_query(const float* __restrict__ z, const float* __restrict__ Wq,
             const float* __restrict__ bq, _Float16* __restrict__ Qh) {
  int i = blockIdx.x * blockDim.x + threadIdx.x;
  if (i >= BATCH * D_OUT) return;
  int b = i / D_OUT, d = i % D_OUT;
  const float* zp = z + (size_t)b * N_LATENT;
  float q = bq[d];
  for (int k = 0; k < N_LATENT; ++k) q += zp[k] * Wq[k * D_OUT + d];
  Qh[i] = (_Float16)q;
}

__global__ __launch_bounds__(256)
void k_temb_cemb(const float* __restrict__ t, const float* __restrict__ c,
                 const float* __restrict__ Wt1, const float* __restrict__ bt1,
                 const float* __restrict__ Wt2, const float* __restrict__ bt2,
                 const float* __restrict__ Wc,  const float* __restrict__ bc,
                 float* __restrict__ temb, float* __restrict__ cemb) {
  int i = blockIdx.x * blockDim.x + threadIdx.x;
  if (i >= BATCH * 16) return;
  int b = i / 16, j = i % 16;
  float tv = t[b];
  float acc = bt2[j];
  for (int k = 0; k < 16; ++k) acc += siluf(tv * Wt1[k] + bt1[k]) * Wt2[k * 16 + j];
  temb[i] = acc;
  const float* cp = c + (size_t)b * N_COND;
  float a2 = bc[j];
  for (int k = 0; k < N_COND; ++k) a2 += cp[k] * Wc[k * 16 + j];
  cemb[i] = a2;
}

// ---------------------------------------------------------------- flash cross-attention
// Per wave: 16 batch rows. Per block: 4 waves. The Tensor Data Mover double-buffers
// 32-gene x 128-ch f16 tiles into LDS (issued by wave 0, EXEC-independent; published
// via s_wait_tensorcnt + workgroup barrier) while waves run WMMAs on the live buffer.
__global__ __launch_bounds__(128)
void k_flash(const _Float16* __restrict__ Qh, const _Float16* __restrict__ gfh,
             float* __restrict__ Ctx) {
  __shared__ __align__(16) _Float16 sG[2][32][D_OUT];  // 2 x 8 KB gene tiles
  __shared__ _Float16 sP[4][16][32];                   // 4 KB P staging (per wave)
  const int tid  = threadIdx.x;
  const int w    = tid >> 5;
  const int lane = tid & 31;
  const int ln16 = lane & 15;
  const int hi   = lane >> 4;
  const int r0   = blockIdx.x * 64 + w * 16;
  const float scale = 0.0883883476483184f;             // 1/sqrt(128)
  const int NT = N_GENES / 32;                         // 625 tiles

  // Preload Q A-operands: 4 K-tiles of 32; A map: j -> k = hi*8+j (j<8) else 16+hi*8+j-8
  v16h a[4];
  for (int kt = 0; kt < 4; ++kt)
    for (int j = 0; j < 16; ++j) {
      int kl = (j < 8) ? (hi * 8 + j) : (16 + hi * 8 + (j - 8));
      a[kt][j] = Qh[(size_t)(r0 + ln16) * D_OUT + kt * 32 + kl];
    }

  float rm[8], rl[8];
  v8f acc[8];
  for (int e = 0; e < 8; ++e) { rm[e] = -INFINITY; rl[e] = 0.f; }
  for (int cc = 0; cc < 8; ++cc)
    for (int e = 0; e < 8; ++e) acc[cc][e] = 0.f;

  // prime the pipeline: TDM load of tile 0 into buffer 0
  if (w == 0) tdm_load_tile((unsigned)(uintptr_t)&sG[0][0][0], gfh);
  __builtin_amdgcn_s_wait_tensorcnt(0);
  __syncthreads();

  for (int it = 0; it < NT; ++it) {
    const int cur = it & 1;
    // kick off async DMA of the NEXT tile into the other buffer (overlaps compute)
    if (w == 0 && it + 1 < NT)
      tdm_load_tile((unsigned)(uintptr_t)&sG[cur ^ 1][0][0],
                    gfh + (size_t)(it + 1) * 32 * D_OUT);

    // ---- scores: two 16x16 D tiles over the 32 genes
    v8f S[2];
    for (int nh = 0; nh < 2; ++nh) {
      v8f s;
      for (int j = 0; j < 8; ++j) s[j] = 0.f;
      const int gl = nh * 16 + ln16;              // B operand: lane = gene column
      for (int kt = 0; kt < 4; ++kt) {
        v16h b;
        for (int j = 0; j < 16; ++j) b[j] = sG[cur][gl][kt * 32 + ((j < 8) ? (hi * 8 + j) : (16 + hi * 8 + (j - 8)))];
        s = __builtin_amdgcn_wmma_f32_16x16x32_f16(false, a[kt], false, b,
                                                   (short)0, s, false, false);
      }
      for (int j = 0; j < 8; ++j) S[nh][j] = s[j] * scale;
    }

    // ---- online softmax bookkeeping (per-row stats live in the 16-lane D groups)
    float tmax[8], csc[8], psum[8];
    for (int e = 0; e < 8; ++e) tmax[e] = fmaxf(S[0][e], S[1][e]);
    for (int off = 1; off < 16; off <<= 1)
      for (int e = 0; e < 8; ++e) tmax[e] = fmaxf(tmax[e], __shfl_xor(tmax[e], off, 16));
    for (int e = 0; e < 8; ++e) {
      float mn = fmaxf(rm[e], tmax[e]);
      csc[e] = __expf(rm[e] - mn);   // 0 on first tile (rm = -inf)
      rm[e] = mn;
    }
    for (int nh = 0; nh < 2; ++nh)
      for (int e = 0; e < 8; ++e) S[nh][e] = __expf(S[nh][e] - rm[e]);
    for (int e = 0; e < 8; ++e) psum[e] = S[0][e] + S[1][e];
    for (int off = 1; off < 16; off <<= 1)
      for (int e = 0; e < 8; ++e) psum[e] += __shfl_xor(psum[e], off, 16);
    for (int e = 0; e < 8; ++e) rl[e] = rl[e] * csc[e] + psum[e];
    for (int cc = 0; cc < 8; ++cc)
      for (int e = 0; e < 8; ++e) acc[cc][e] *= csc[e];

    // ---- transpose P (D layout -> row-major) through per-wave LDS staging
    for (int nh = 0; nh < 2; ++nh)
      for (int e = 0; e < 8; ++e)
        sP[w][hi * 8 + e][nh * 16 + ln16] = (_Float16)S[nh][e];
    // same-wave DS ops are in-order; explicit wait keeps the RAW safe without a barrier
    asm volatile("s_wait_dscnt 0" ::: "memory");

    v16h pa;   // A operand for P: lane = batch row, k = gene-in-tile
    for (int j = 0; j < 16; ++j) {
      int kl = (j < 8) ? (hi * 8 + j) : (16 + hi * 8 + (j - 8));
      pa[j] = sP[w][ln16][kl];
    }
    // ---- context accumulate: 8 channel chunks, B operand: lane = channel, k = gene
    for (int cc = 0; cc < 8; ++cc) {
      v16h b;
      const int n = cc * 16 + ln16;
      for (int j = 0; j < 16; ++j) b[j] = sG[cur][(j < 8) ? (hi * 8 + j) : (16 + hi * 8 + (j - 8))][n];
      acc[cc] = __builtin_amdgcn_wmma_f32_16x16x32_f16(false, pa, false, b,
                                                       (short)0, acc[cc], false, false);
    }

    // publish the DMA'd next tile to the whole block before flipping buffers
    __builtin_amdgcn_s_wait_tensorcnt(0);
    __syncthreads();
  }

  // ---- normalize and emit context
  for (int cc = 0; cc < 8; ++cc)
    for (int e = 0; e < 8; ++e)
      Ctx[(size_t)(r0 + hi * 8 + e) * D_OUT + cc * 16 + ln16] = acc[cc][e] / rl[e];
}

// ---------------------------------------------------------------- fused MLP epilogue
__global__ __launch_bounds__(128)
void k_mlp(const float* __restrict__ z, const float* __restrict__ Ctx,
           const float* __restrict__ temb, const float* __restrict__ cemb,
           const float* __restrict__ Wf1, const float* __restrict__ bf1,
           const float* __restrict__ Wf2, const float* __restrict__ bf2,
           float* __restrict__ out) {
  __shared__ float vin[N_LATENT + D_OUT + 32];  // 210
  __shared__ float hbuf[128];
  const int b = blockIdx.x, tid = threadIdx.x;
  for (int i = tid; i < 210; i += 128) {
    float v;
    if (i < 50)        v = z[(size_t)b * N_LATENT + i];
    else if (i < 178)  v = Ctx[(size_t)b * D_OUT + (i - 50)];
    else if (i < 194)  v = temb[(size_t)b * 16 + (i - 178)];
    else               v = cemb[(size_t)b * 16 + (i - 194)];
    vin[i] = v;
  }
  __syncthreads();
  float acc = bf1[tid];
  for (int k = 0; k < 210; ++k) acc += vin[k] * Wf1[k * 128 + tid];
  hbuf[tid] = siluf(acc);
  __syncthreads();
  if (tid < N_LATENT) {
    float o = bf2[tid];
    for (int k = 0; k < 128; ++k) o += hbuf[k] * Wf2[k * N_LATENT + tid];
    out[(size_t)b * N_LATENT + tid] = o;
  }
}

// ---------------------------------------------------------------- launcher
extern "C" void kernel_launch(void* const* d_in, const int* in_sizes, int n_in,
                              void* d_out, int out_size, void* d_ws, size_t ws_size,
                              hipStream_t stream) {
  const float* t     = (const float*)d_in[0];
  const float* z     = (const float*)d_in[1];
  const float* c     = (const float*)d_in[2];
  const int*   ei    = (const int*)  d_in[3];
  const float* ge    = (const float*)d_in[4];
  const float* Wl    = (const float*)d_in[5];
  const float* bl    = (const float*)d_in[6];
  const float* Wr    = (const float*)d_in[7];
  const float* br    = (const float*)d_in[8];
  const float* att   = (const float*)d_in[9];
  const float* gbias = (const float*)d_in[10];
  const float* Wq    = (const float*)d_in[11];
  const float* bq    = (const float*)d_in[12];
  const float* Wt1   = (const float*)d_in[13];
  const float* bt1   = (const float*)d_in[14];
  const float* Wt2   = (const float*)d_in[15];
  const float* bt2   = (const float*)d_in[16];
  const float* Wc    = (const float*)d_in[17];
  const float* bc    = (const float*)d_in[18];
  const float* Wf1   = (const float*)d_in[19];
  const float* bf1   = (const float*)d_in[20];
  const float* Wf2   = (const float*)d_in[21];
  const float* bf2   = (const float*)d_in[22];
  float* out = (float*)d_out;

  char* wsb = (char*)d_ws;
  size_t off = 0;
  auto alloc = [&](size_t bytes) -> char* {
    char* p = wsb + off;
    off = (off + bytes + 255) & ~(size_t)255;
    return p;
  };
  float*    xl     = (float*)   alloc((size_t)N_GENES * D_OUT * 4);
  float*    xr     = (float*)   alloc((size_t)N_GENES * D_OUT * 4);
  float*    accum  = (float*)   alloc((size_t)N_GENES * D_OUT * 4);
  float*    escore = (float*)   alloc((size_t)E_ALL * HEADS * 4);
  float*    emax   = (float*)   alloc((size_t)N_GENES * HEADS * 4);
  float*    den    = (float*)   alloc((size_t)N_GENES * HEADS * 4);
  _Float16* gfh    = (_Float16*)alloc((size_t)N_GENES * D_OUT * 2);
  _Float16* Qh     = (_Float16*)alloc((size_t)BATCH * D_OUT * 2);
  float*    Ctx    = (float*)   alloc((size_t)BATCH * D_OUT * 4);
  float*    temb   = (float*)   alloc((size_t)BATCH * 16 * 4);
  float*    cemb   = (float*)   alloc((size_t)BATCH * 16 * 4);

  const int ND = N_GENES * D_OUT;           // 2,560,000
  const int EH = E_ALL * HEADS;             // 1,320,000

  k_node_transform<<<(ND + 255) / 256, 256, 0, stream>>>(ge, Wl, bl, Wr, br, xl, xr);
  k_init<<<(ND + 255) / 256, 256, 0, stream>>>(accum, emax, den);
  k_edge_score<<<(EH + 255) / 256, 256, 0, stream>>>(ei, xl, xr, att, escore, emax);
  k_edge_accum<<<(EH + 255) / 256, 256, 0, stream>>>(ei, xl, escore, emax, den, accum);
  k_finalize<<<(ND + 255) / 256, 256, 0, stream>>>(accum, den, gbias, gfh);
  k_query<<<(BATCH * D_OUT + 255) / 256, 256, 0, stream>>>(z, Wq, bq, Qh);
  k_temb_cemb<<<(BATCH * 16 + 255) / 256, 256, 0, stream>>>(t, c, Wt1, bt1, Wt2, bt2,
                                                            Wc, bc, temb, cemb);
  k_flash<<<BATCH / 64, 128, 0, stream>>>(Qh, gfh, Ctx);
  k_mlp<<<BATCH, 128, 0, stream>>>(z, Ctx, temb, cemb, Wf1, bf1, Wf2, bf2, out);
}
